// Model_13391708029675
// MI455X (gfx1250) — matrix operand order repaired and checked by reading, not compiled, not run on hardware
//
#include <hip/hip_runtime.h>
#include <stdint.h>

// ---------- types for WMMA ----------
typedef __attribute__((ext_vector_type(16))) __bf16         v16bf;
typedef __attribute__((ext_vector_type(8)))  float          v8f;
typedef __attribute__((ext_vector_type(8)))  unsigned short us8;

struct BF16Frag { us8 lo, hi; };   // 32 bytes == v16bf

static inline int cdiv_h(long long a, long long b) { return (int)((a + b - 1) / b); }

// ---------- monotonic float<->uint encoding for atomic max ----------
__device__ __forceinline__ unsigned fenc(float f) {
  unsigned u = __float_as_uint(f);
  return (u & 0x80000000u) ? ~u : (u | 0x80000000u);
}
__device__ __forceinline__ float fdec(unsigned u) {
  return (u & 0x80000000u) ? __uint_as_float(u & 0x7FFFFFFFu) : __uint_as_float(~u);
}

// ---------- utility kernels ----------
__global__ void k_zero(uint32_t* p, long long n) {
  long long i = (long long)blockIdx.x * blockDim.x + threadIdx.x;
  if (i < n) p[i] = 0u;
}

__global__ void k_f32_to_bf16(const float* __restrict__ in, unsigned short* __restrict__ out,
                              long long n) {
  long long i = (long long)blockIdx.x * blockDim.x + threadIdx.x;
  if (i >= n) return;
  unsigned b = __float_as_uint(in[i]);
  unsigned r = (b + 0x7FFFu + ((b >> 16) & 1u)) >> 16;   // RNE
  out[i] = (unsigned short)r;
}

// degree + sum of incoming edge_attr (for self-loop fill_value='mean')
__global__ void k_deg_loop(const int* __restrict__ dstI, const float* __restrict__ eattr,
                           float* __restrict__ deg, float* __restrict__ loop_sum, int E) {
  int e = blockIdx.x * blockDim.x + threadIdx.x;
  if (e >= E) return;
  int d = dstI[e];
  atomicAdd(&deg[d], 1.0f);
  const float* a = eattr + (size_t)e * 32;
  float* o = loop_sum + (size_t)d * 32;
  #pragma unroll
  for (int c = 0; c < 32; c++) atomicAdd(&o[c], a[c]);
}

__global__ void k_loop_div(float* __restrict__ loop_attr, const float* __restrict__ deg, int N) {
  int i = blockIdx.x * blockDim.x + threadIdx.x;
  if (i >= N * 32) return;
  loop_attr[i] = loop_attr[i] / fmaxf(deg[i >> 5], 1.0f);
}

// ---------- WMMA bf16 GEMM:  out[Nr,M] = A[Nr,K] (bf16) @ W[M,K]^T (bf16), f32 accum ----------
// One wave32 computes a 16x64 output strip: 4 WMMA tiles sharing one A fragment per K-step.
// Fragment layouts used here (contiguous k, same order for A and B):
//   A (16x32 bf16): lane row m=lane&15; elems 0..15 -> k = kb + 16*(lane>>4) + i
//   B (32x16 bf16): lane col n=lane&15; elems 0..15 -> k = kb + 16*(lane>>4) + i
//   C/D (16x16 f32): lane col n=lane&15; vgpr j -> row m = j + 8*(lane>>4)
// A rows are clamped (not predicated) so loads are unconditional; stores are guarded.
__global__ void k_gemm_bf16(const unsigned short* __restrict__ A,
                            const unsigned short* __restrict__ W,
                            float* __restrict__ out,
                            int Nr, int K, int M, int totalStrips) {
  int wave = (int)(((long long)blockIdx.x * blockDim.x + threadIdx.x) >> 5);
  int lane = threadIdx.x & 31;
  if (wave >= totalStrips) return;
  int grpM = M >> 6;                      // 64-wide column groups
  int rt = wave / grpM, cg = wave % grpM;
  int row = rt << 4, col = cg << 6;
  int n16  = lane & 15;
  int hi   = lane >> 4;
  int mCl  = row + n16; if (mCl > Nr - 1) mCl = Nr - 1;   // clamped, always valid
  const unsigned short* ap = A + (size_t)mCl * K + (hi << 4);
  const unsigned short* bp = W + (size_t)(col + n16) * K + (hi << 4);
  const size_t wStep = (size_t)16 * K;    // 16 weight rows per column tile
  v8f c0 = {}, c1 = {}, c2 = {}, c3 = {};
  for (int kb = 0; kb < K; kb += 32) {
    BF16Frag fa;
    fa.lo = *(const us8*)(ap + kb);
    fa.hi = *(const us8*)(ap + kb + 8);
    v16bf a = __builtin_bit_cast(v16bf, fa);
    BF16Frag fb;
    fb.lo = *(const us8*)(bp + kb);
    fb.hi = *(const us8*)(bp + kb + 8);
    c0 = __builtin_amdgcn_wmma_f32_16x16x32_bf16(false, a, false,
         __builtin_bit_cast(v16bf, fb), (short)0, c0, false, false);
    fb.lo = *(const us8*)(bp + wStep + kb);
    fb.hi = *(const us8*)(bp + wStep + kb + 8);
    c1 = __builtin_amdgcn_wmma_f32_16x16x32_bf16(false, a, false,
         __builtin_bit_cast(v16bf, fb), (short)0, c1, false, false);
    fb.lo = *(const us8*)(bp + 2 * wStep + kb);
    fb.hi = *(const us8*)(bp + 2 * wStep + kb + 8);
    c2 = __builtin_amdgcn_wmma_f32_16x16x32_bf16(false, a, false,
         __builtin_bit_cast(v16bf, fb), (short)0, c2, false, false);
    fb.lo = *(const us8*)(bp + 3 * wStep + kb);
    fb.hi = *(const us8*)(bp + 3 * wStep + kb + 8);
    c3 = __builtin_amdgcn_wmma_f32_16x16x32_bf16(false, a, false,
         __builtin_bit_cast(v16bf, fb), (short)0, c3, false, false);
  }
  #pragma unroll
  for (int j = 0; j < 8; j++) {
    int mr = row + j + (hi << 3);
    if (mr < Nr) {
      float* op = out + (size_t)mr * M + col + n16;
      op[0]  = c0[j];
      op[16] = c1[j];
      op[32] = c2[j];
      op[48] = c3[j];
    }
  }
}

// ---------- attention vectors folded into edge weights: u[h,d] = sum_c att[h,c]*lin_edge[h*C+c,d] ----------
__global__ void k_uvec(const float* __restrict__ w1_le, const float* __restrict__ a1e,
                       const float* __restrict__ w2_le, const float* __restrict__ a2e,
                       float* __restrict__ u1, float* __restrict__ u2) {
  int i = blockIdx.x * blockDim.x + threadIdx.x;
  if (i < 128) {
    int h = i >> 5, d = i & 31;
    float s = 0.f;
    for (int c = 0; c < 64; c++) s += a1e[h * 64 + c] * w1_le[(h * 64 + c) * 32 + d];
    u1[i] = s;
  } else if (i < 160) {
    int d = i - 128;
    float s = 0.f;
    for (int c = 0; c < 64; c++) s += a2e[c] * w2_le[c * 32 + d];
    u2[d] = s;
  }
}

// ---------- per-node attention scalars ----------
__global__ void k_att_node1(const float* __restrict__ xl, const float* __restrict__ att_src,
                            const float* __restrict__ att_dst,
                            float* __restrict__ as, float* __restrict__ ad, int N) {
  int i = blockIdx.x * blockDim.x + threadIdx.x;
  if (i >= N * 4) return;
  int n = i >> 2, h = i & 3;
  const float* xp = xl + (size_t)n * 256 + h * 64;
  const float* ps = att_src + h * 64;
  const float* pd = att_dst + h * 64;
  float s = 0.f, d = 0.f;
  for (int c = 0; c < 64; c++) { float v = xp[c]; s += v * ps[c]; d += v * pd[c]; }
  as[i] = s; ad[i] = d;
}

__global__ void k_att_node2(const float* __restrict__ h2l, const float* __restrict__ att_src,
                            const float* __restrict__ att_dst,
                            float* __restrict__ as, float* __restrict__ ad, int N) {
  int n = blockIdx.x * blockDim.x + threadIdx.x;
  if (n >= N) return;
  const float* xp = h2l + (size_t)n * 64;
  float s = 0.f, d = 0.f;
  for (int c = 0; c < 64; c++) { float v = xp[c]; s += v * att_src[c]; d += v * att_dst[c]; }
  as[n] = s; ad[n] = d;
}

// ---------- edge pass 1: alpha = leaky_relu(as[src]+ad[dst]+a_edge), atomic segment max ----------
__global__ void k_edge_alpha1(const int* __restrict__ srcI, const int* __restrict__ dstI,
                              const float* __restrict__ eattr, const float* __restrict__ loop_attr,
                              const float* __restrict__ u1,
                              const float* __restrict__ as, const float* __restrict__ ad,
                              float* __restrict__ alpha, unsigned* __restrict__ mU, int E, int N) {
  int e = blockIdx.x * blockDim.x + threadIdx.x;
  if (e >= E + N) return;
  int s, d; const float* attr;
  if (e < E) { s = srcI[e]; d = dstI[e]; attr = eattr + (size_t)e * 32; }
  else       { s = d = e - E;            attr = loop_attr + (size_t)s * 32; }
  float h0 = 0.f, h1 = 0.f, h2 = 0.f, h3 = 0.f;
  for (int c = 0; c < 32; c++) {
    float a = attr[c];
    h0 += a * u1[c]; h1 += a * u1[32 + c]; h2 += a * u1[64 + c]; h3 += a * u1[96 + c];
  }
  float ae[4] = { h0, h1, h2, h3 };
  #pragma unroll
  for (int h = 0; h < 4; h++) {
    float al = as[s * 4 + h] + ad[d * 4 + h] + ae[h];
    al = (al > 0.f) ? al : 0.2f * al;
    alpha[(size_t)e * 4 + h] = al;
    atomicMax(&mU[d * 4 + h], fenc(al));
  }
}

__global__ void k_edge_alpha2(const int* __restrict__ srcI, const int* __restrict__ dstI,
                              const float* __restrict__ eattr, const float* __restrict__ loop_attr,
                              const float* __restrict__ u2,
                              const float* __restrict__ as, const float* __restrict__ ad,
                              float* __restrict__ alpha, unsigned* __restrict__ mU, int E, int N) {
  int e = blockIdx.x * blockDim.x + threadIdx.x;
  if (e >= E + N) return;
  int s, d; const float* attr;
  if (e < E) { s = srcI[e]; d = dstI[e]; attr = eattr + (size_t)e * 32; }
  else       { s = d = e - E;            attr = loop_attr + (size_t)s * 32; }
  float ae = 0.f;
  for (int c = 0; c < 32; c++) ae += attr[c] * u2[c];
  float al = as[s] + ad[d] + ae;
  al = (al > 0.f) ? al : 0.2f * al;
  alpha[e] = al;
  atomicMax(&mU[d], fenc(al));
}

// ---------- edge pass 2: ex = exp(alpha - max); segment-sum denom ----------
__global__ void k_edge_exp(const int* __restrict__ dstI, float* __restrict__ alpha,
                           const unsigned* __restrict__ mU, float* __restrict__ denom,
                           int E, int N, int H) {
  long long i = (long long)blockIdx.x * blockDim.x + threadIdx.x;
  long long tot = (long long)(E + N) * H;
  if (i >= tot) return;
  int e = (int)(i / H), h = (int)(i % H);
  int d = (e < E) ? dstI[e] : (e - E);
  float ex = __expf(alpha[i] - fdec(mU[d * H + h]));
  alpha[i] = ex;
  atomicAdd(&denom[d * H + h], ex);
}

// ---------- edge pass 3: agg[dst] += feat[src] * ex/denom ; H*C/4 lanes per edge, float4 gather ----------
__global__ void k_aggregate(const int* __restrict__ srcI, const int* __restrict__ dstI,
                            const float* __restrict__ feat, const float* __restrict__ ex,
                            const float* __restrict__ denom, float* __restrict__ agg,
                            int E, int N, int H, int C) {
  int tpe = (H * C) >> 2;
  long long gid = (long long)blockIdx.x * blockDim.x + threadIdx.x;
  int e = (int)(gid / tpe);
  int t = (int)(gid % tpe);
  if (e >= E + N) return;
  int s, d;
  if (e < E) { s = srcI[e]; d = dstI[e]; } else { s = d = e - E; }
  int h = (t * 4) / C;
  float w = ex[(size_t)e * H + h] / (denom[(size_t)d * H + h] + 1e-16f);
  const float* fp = feat + (size_t)s * (H * C) + t * 4;
  float*       op = agg  + (size_t)d * (H * C) + t * 4;
  float4 v = *(const float4*)fp;
  atomicAdd(&op[0], v.x * w);
  atomicAdd(&op[1], v.y * w);
  atomicAdd(&op[2], v.z * w);
  atomicAdd(&op[3], v.w * w);
}

// ---------- layer-1 epilogue: h = relu(agg + bias) -> bf16 for layer-2 WMMA ----------
__global__ void k_relu_bias_bf16(const float* __restrict__ agg, const float* __restrict__ bias,
                                 unsigned short* __restrict__ out, long long NC, int C) {
  long long i = (long long)blockIdx.x * blockDim.x + threadIdx.x;
  if (i >= NC) return;
  float v = agg[i] + bias[(int)(i % C)];
  v = fmaxf(v, 0.f);
  unsigned b = __float_as_uint(v);
  unsigned r = (b + 0x7FFFu + ((b >> 16) & 1u)) >> 16;
  out[i] = (unsigned short)r;
}

// ---------- final: out = agg2 + bias2 ----------
__global__ void k_final(const float* __restrict__ agg, const float* __restrict__ bias,
                        float* __restrict__ out, long long NC) {
  long long i = (long long)blockIdx.x * blockDim.x + threadIdx.x;
  if (i >= NC) return;
  out[i] = agg[i] + bias[(int)(i & 63)];
}

// ======================= host launch =======================
extern "C" void kernel_launch(void* const* d_in, const int* in_sizes, int n_in,
                              void* d_out, int out_size, void* d_ws, size_t ws_size,
                              hipStream_t stream) {
  (void)n_in; (void)out_size; (void)ws_size;
  const float* x           = (const float*)d_in[0];
  const int*   eidx        = (const int*)  d_in[1];
  const float* eattr       = (const float*)d_in[2];
  const float* w1_lin      = (const float*)d_in[3];
  const float* w1_att_src  = (const float*)d_in[4];
  const float* w1_att_dst  = (const float*)d_in[5];
  const float* w1_lin_edge = (const float*)d_in[6];
  const float* w1_att_edge = (const float*)d_in[7];
  const float* w1_bias     = (const float*)d_in[8];
  const float* w2_lin      = (const float*)d_in[9];
  const float* w2_att_src  = (const float*)d_in[10];
  const float* w2_att_dst  = (const float*)d_in[11];
  const float* w2_lin_edge = (const float*)d_in[12];
  const float* w2_att_edge = (const float*)d_in[13];
  const float* w2_bias     = (const float*)d_in[14];

  const int N  = in_sizes[0] / 128;
  const int E  = in_sizes[2] / 32;
  const int E2 = E + N;
  const int* srcI = eidx;
  const int* dstI = eidx + E;

  // ---- workspace carve (zero-needed block first, zeroed once per call) ----
  char* p = (char*)d_ws;
  auto carve = [&](size_t bytes) -> char* {
    char* r = p; p += (bytes + 255) & ~(size_t)255; return r;
  };
  float*    deg       = (float*)   carve((size_t)N * 4);
  float*    loop_attr = (float*)   carve((size_t)N * 32 * 4);
  unsigned* m1        = (unsigned*)carve((size_t)N * 4 * 4);
  float*    den1      = (float*)   carve((size_t)N * 4 * 4);
  float*    agg1      = (float*)   carve((size_t)N * 256 * 4);
  unsigned* m2        = (unsigned*)carve((size_t)N * 4);
  float*    den2      = (float*)   carve((size_t)N * 4);
  float*    agg2      = (float*)   carve((size_t)N * 64 * 4);
  const long long zeroWords = (long long)(p - (char*)d_ws) / 4;
  unsigned short* xb  = (unsigned short*)carve((size_t)N * 128 * 2);
  unsigned short* w1b = (unsigned short*)carve((size_t)256 * 128 * 2);
  unsigned short* w2b = (unsigned short*)carve((size_t)64 * 256 * 2);
  float*    xl        = (float*)   carve((size_t)N * 256 * 4);
  float*    u1        = (float*)   carve(128 * 4);
  float*    u2        = (float*)   carve(32 * 4);
  float*    as1       = (float*)   carve((size_t)N * 4 * 4);
  float*    ad1       = (float*)   carve((size_t)N * 4 * 4);
  float*    alpha1    = (float*)   carve((size_t)E2 * 4 * 4);
  unsigned short* hb  = (unsigned short*)carve((size_t)N * 256 * 2);
  float*    h2l       = (float*)   carve((size_t)N * 64 * 4);
  float*    as2       = (float*)   carve((size_t)N * 4);
  float*    ad2       = (float*)   carve((size_t)N * 4);
  float*    alpha2    = (float*)   carve((size_t)E2 * 4);

  const int B = 256;
  // 0) zero accumulators
  k_zero<<<cdiv_h(zeroWords, B), B, 0, stream>>>((uint32_t*)d_ws, zeroWords);
  // 1) self-loop attr = mean of incoming edge_attr
  k_deg_loop<<<cdiv_h(E, B), B, 0, stream>>>(dstI, eattr, deg, loop_attr, E);
  k_loop_div<<<cdiv_h((long long)N * 32, B), B, 0, stream>>>(loop_attr, deg, N);
  // 2) bf16 conversions for WMMA inputs
  k_f32_to_bf16<<<cdiv_h((long long)N * 128, B), B, 0, stream>>>(x, xb, (long long)N * 128);
  k_f32_to_bf16<<<cdiv_h(256 * 128, B), B, 0, stream>>>(w1_lin, w1b, 256 * 128);
  k_f32_to_bf16<<<cdiv_h(64 * 256, B), B, 0, stream>>>(w2_lin, w2b, 64 * 256);
  // 3) xl = x @ W1^T  (WMMA bf16, 16x64 strips)
  const int strips1 = cdiv_h(N, 16) * (256 / 64);
  k_gemm_bf16<<<cdiv_h((long long)strips1 * 32, B), B, 0, stream>>>(xb, w1b, xl, N, 128, 256, strips1);
  // 4) folded edge-attention vectors
  k_uvec<<<1, B, 0, stream>>>(w1_lin_edge, w1_att_edge, w2_lin_edge, w2_att_edge, u1, u2);
  // 5) layer 1 attention + aggregation
  k_att_node1<<<cdiv_h((long long)N * 4, B), B, 0, stream>>>(xl, w1_att_src, w1_att_dst, as1, ad1, N);
  k_edge_alpha1<<<cdiv_h(E2, B), B, 0, stream>>>(srcI, dstI, eattr, loop_attr, u1, as1, ad1, alpha1, m1, E, N);
  k_edge_exp<<<cdiv_h((long long)E2 * 4, B), B, 0, stream>>>(dstI, alpha1, m1, den1, E, N, 4);
  k_aggregate<<<cdiv_h((long long)E2 * 64, B), B, 0, stream>>>(srcI, dstI, xl, alpha1, den1, agg1, E, N, 4, 64);
  // 6) h = relu(agg1 + b1) -> bf16
  k_relu_bias_bf16<<<cdiv_h((long long)N * 256, B), B, 0, stream>>>(agg1, w1_bias, hb, (long long)N * 256, 256);
  // 7) h2l = h @ W2^T (WMMA bf16, 16x64 strips)
  const int strips2 = cdiv_h(N, 16) * (64 / 64);
  k_gemm_bf16<<<cdiv_h((long long)strips2 * 32, B), B, 0, stream>>>(hb, w2b, h2l, N, 256, 64, strips2);
  // 8) layer 2 attention + aggregation
  k_att_node2<<<cdiv_h(N, B), B, 0, stream>>>(h2l, w2_att_src, w2_att_dst, as2, ad2, N);
  k_edge_alpha2<<<cdiv_h(E2, B), B, 0, stream>>>(srcI, dstI, eattr, loop_attr, u2, as2, ad2, alpha2, m2, E, N);
  k_edge_exp<<<cdiv_h((long long)E2, B), B, 0, stream>>>(dstI, alpha2, m2, den2, E, N, 1);
  k_aggregate<<<cdiv_h((long long)E2 * 16, B), B, 0, stream>>>(srcI, dstI, h2l, alpha2, den2, agg2, E, N, 1, 64);
  // 9) out = agg2 + b2
  k_final<<<cdiv_h((long long)N * 64, B), B, 0, stream>>>(agg2, w2_bias, (float*)d_out, (long long)N * 64);
}
